// LocalMHA_85392539779218
// MI455X (gfx1250) — matrix operand order repaired and checked by reading, not compiled, not run on hardware
//
#include <hip/hip_runtime.h>
#include <hip/hip_bf16.h>

typedef __attribute__((ext_vector_type(16))) _Float16 v16h;
typedef __attribute__((ext_vector_type(8)))  _Float16 v8h;
typedef __attribute__((ext_vector_type(8)))  float    v8f;

#define HEADS 8
#define DHEAD 64
#define NSEQ  8192
#define BATCH 4
#define DIM   512
#define INNER 512
#define WIN   128
#define NW    (NSEQ / WIN)         // 64
#define MTOT  (BATCH * NSEQ)       // 32768

// ---------------- GEMM tiles ----------------
#define TM 128
#define TN 128
#define TK 32
#define A_LD 40                    // halves; 80B row stride (16B aligned)
// B tiles in fragment-major layout: [k/16][col][k%16], col stride padded
#define BSW_NLD 24                 // 48B per col: 16B aligned, 12-dword bank stride
#define BSW_CH  (TN * BSW_NLD)
#define BSW_SZ  (2 * BSW_CH)

// ---- CDNA5 async global->LDS copy (ASYNCcnt-tracked DMA, 16B per lane) ----
static __device__ __forceinline__ void async_copy16(void* lds_dst, const void* gsrc) {
  // generic LDS pointer: addr[31:0] is the LDS byte offset (ISA 10.2 aperture rule)
  unsigned ldsoff = (unsigned)(size_t)lds_dst;
  asm volatile("global_load_async_to_lds_b128 %0, %1, off"
               :: "v"(ldsoff), "v"(gsrc) : "memory");
}
static __device__ __forceinline__ void wait_async0() {
  asm volatile("s_wait_asynccnt 0" ::: "memory");
}

// concat two contiguous 8-half groups into a WMMA fragment (2x ds_load_b128)
static __device__ __forceinline__ v16h frag_cat(const _Float16* p0, const _Float16* p1) {
  v8h a = *(const v8h*)p0, b = *(const v8h*)p1;
  return __builtin_shufflevector(a, b, 0,1,2,3,4,5,6,7,8,9,10,11,12,13,14,15);
}

// A (16x32 f16): M = lane%16, element e -> K = half*16 + e
static __device__ __forceinline__ v16h a_frag(const _Float16* base, int ld,
                                              int row, int k0, int half) {
  const _Float16* p = base + row * ld + k0 + half * 16;
  return frag_cat(p, p + 8);
}

// -------- staging helpers (256-thread blocks) --------
static __device__ __forceinline__ void stage_a_f32(_Float16* As_, const float* X,
                                                   int ldx, int m0, int k0, int t) {
#pragma unroll
  for (int i = 0; i < 4; ++i) {                 // 128x32 f32 -> f16
    int id = t + 256 * i;
    int r = id >> 3, c4 = (id & 7) << 2;
    const float4 f = *(const float4*)(X + (size_t)(m0 + r) * ldx + k0 + c4);
    _Float16* d = &As_[r * A_LD + c4];
    d[0] = (_Float16)f.x; d[1] = (_Float16)f.y;
    d[2] = (_Float16)f.z; d[3] = (_Float16)f.w;
  }
}
static __device__ __forceinline__ void stage_a_f16_async(_Float16* As_, const _Float16* Ah,
                                                         int lda, int m0, int k0, int t) {
#pragma unroll
  for (int i = 0; i < 2; ++i) {                 // 128x32 halves via async DMA
    int id = t + 256 * i;
    int r = id >> 2, p = id & 3;
    async_copy16(&As_[r * A_LD + p * 8],
                 &Ah[(size_t)(m0 + r) * lda + k0 + p * 8]);
  }
}
static __device__ __forceinline__ void stage_b_f32(_Float16* Bs_, const float* W,
                                                   int ldw, int k0, int n0, int t) {
#pragma unroll
  for (int i = 0; i < 4; ++i) {                 // 32x128 f32 -> fragment-major f16
    int id = t + 256 * i;
    int r = id >> 5, c4 = (id & 31) << 2;       // r = k in tile, c4 = col
    const float4 f = *(const float4*)(W + (size_t)(k0 + r) * ldw + n0 + c4);
    _Float16* d = &Bs_[(r >> 4) * BSW_CH + c4 * BSW_NLD + (r & 15)];
    d[0 * BSW_NLD] = (_Float16)f.x; d[1 * BSW_NLD] = (_Float16)f.y;
    d[2 * BSW_NLD] = (_Float16)f.z; d[3 * BSW_NLD] = (_Float16)f.w;
  }
}

#define WMMA_F16(A, B, C) \
  __builtin_amdgcn_wmma_f32_16x16x32_f16(false, (A), false, (B), (short)0, (C), false, false)

// ================= Kernel 1: fused QKV projection =================
__global__ __launch_bounds__(256)
void qkv_gemm(const float* __restrict__ X, const float* __restrict__ Wqkv,
              _Float16* __restrict__ qh, _Float16* __restrict__ kh,
              _Float16* __restrict__ vh) {
  __shared__ _Float16 As[2][TM * A_LD];
  __shared__ _Float16 Bs[2][BSW_SZ];
  const int t = threadIdx.x;
  const int wv = t >> 5, lane = t & 31, half = lane >> 4, ml = lane & 15;
  const int m0 = blockIdx.y * TM;
  const int n0 = blockIdx.x * TN;
  const int NK = DIM / TK;

  v8f acc[8];
#pragma unroll
  for (int i = 0; i < 8; ++i) acc[i] = (v8f)0.f;

  stage_a_f32(As[0], X, DIM, m0, 0, t);
  stage_b_f32(Bs[0], Wqkv, 3 * INNER, 0, n0, t);
  __syncthreads();

  for (int kt = 0; kt < NK; ++kt) {
    const int cur = kt & 1;
    if (kt + 1 < NK) {
      stage_a_f32(As[cur ^ 1], X, DIM, m0, (kt + 1) * TK, t);
      stage_b_f32(Bs[cur ^ 1], Wqkv, 3 * INNER, (kt + 1) * TK, n0, t);
      if (kt + 2 < NK) {
        __builtin_prefetch(X + (size_t)(m0 + (t >> 1)) * DIM + (kt + 2) * TK, 0, 3);
        __builtin_prefetch(Wqkv + (size_t)((kt + 2) * TK + (t & 31)) * (3 * INNER) + n0, 0, 3);
      }
    }
    // batch all fragment loads, then burst the WMMAs
    v16h a = a_frag(As[cur], A_LD, wv * 16 + ml, 0, half);
    v16h bf[8];
#pragma unroll
    for (int nt = 0; nt < 8; ++nt) {
      const _Float16* bp = &Bs[cur][half * BSW_CH + (nt * 16 + ml) * BSW_NLD];
      bf[nt] = frag_cat(bp, bp + 8);
    }
#pragma unroll
    for (int nt = 0; nt < 8; ++nt) acc[nt] = WMMA_F16(a, bf[nt], acc[nt]);
    __syncthreads();
  }

  // epilogue: split columns into Q/K/V heads, fold in d^-0.5 on Q
#pragma unroll
  for (int nt = 0; nt < 8; ++nt) {
    int col = n0 + nt * 16 + ml;
    int which = col >> 9;               // 0:Q 1:K 2:V
    int hh = (col >> 6) & 7;
    int dd = col & 63;
    _Float16* dst = (which == 0) ? qh : ((which == 1) ? kh : vh);
    float sc = (which == 0) ? 0.125f : 1.0f;
#pragma unroll
    for (int r = 0; r < 8; ++r) {
      int grow = m0 + wv * 16 + r + half * 8;
      int b = grow >> 13, n = grow & (NSEQ - 1);
      dst[((size_t)(b * HEADS + hh) * NSEQ + n) * DHEAD + dd] =
          (_Float16)(acc[nt][r] * sc);
    }
  }
}

// ================= Kernel 2: windowed local attention =================
#define QS_LD 72
#define PS_LD 392
#define KCH   (384 * 24)   // Ksw: [4 d-chunks][384 keys][24]
#define VCH   (64 * 24)    // Vsw: [24 key-chunks][64 dims][24]
#define SMEM_HALVES (128 * QS_LD + 4 * KCH + 24 * VCH + 8 * 16 * PS_LD)

__global__ __launch_bounds__(256)
void local_attn(const _Float16* __restrict__ qh, const _Float16* __restrict__ kh,
                const _Float16* __restrict__ vh, _Float16* __restrict__ oh) {
  extern __shared__ char smem[];
  _Float16* Qs  = (_Float16*)smem;        // [128][QS_LD] row-major
  _Float16* Ksw = Qs + 128 * QS_LD;       // K^T, fragment-major over head dim
  _Float16* Vsw = Ksw + 4 * KCH;          // V, fragment-major over key dim
  _Float16* Ps  = Vsw + 24 * VCH;         // per-wave [16][PS_LD]

  const int t = threadIdx.x;
  const int wv = t >> 5, lane = t & 31, half = lane >> 4, ml = lane & 15;
  const int win = blockIdx.x, h = blockIdx.y, b = blockIdx.z;
  const size_t basebh = (size_t)(b * HEADS + h) * NSEQ * DHEAD;
  const int qrow0 = win * WIN;
  const int krow0 = win * WIN - WIN;

  // ---- stage Q via async DMA (128x64, row-major) ----
#pragma unroll
  for (int i = 0; i < 4; ++i) {
    int id = t + 256 * i;
    int r = id >> 3, p = id & 7;
    async_copy16(&Qs[r * QS_LD + p * 8],
                 &qh[basebh + (size_t)(qrow0 + r) * DHEAD + p * 8]);
  }
  // ---- stage K (async, lands directly in transposed fragment-major layout)
  //      and V (register path: scatter to fragment-major) ----
#pragma unroll
  for (int i = 0; i < 12; ++i) {
    int id = t + 256 * i;
    int r = id >> 3, p = id & 7;          // r = key row 0..383, p = dim chunk
    int gr = krow0 + r;
    _Float16* kdst = &Ksw[(p >> 1) * KCH + r * 24 + (p & 1) * 8];
    if ((unsigned)gr < (unsigned)NSEQ) {
      async_copy16(kdst, &kh[basebh + (size_t)gr * DHEAD + p * 8]);
      int4 vv = *(const int4*)&vh[basebh + (size_t)gr * DHEAD + p * 8];
      const _Float16* vp = (const _Float16*)&vv;
#pragma unroll
      for (int x = 0; x < 8; ++x)
        Vsw[(r >> 4) * VCH + (p * 8 + x) * 24 + (r & 15)] = vp[x];
    } else {
      int4 z; z.x = z.y = z.z = z.w = 0;
      *(int4*)kdst = z;
#pragma unroll
      for (int x = 0; x < 8; ++x)
        Vsw[(r >> 4) * VCH + (p * 8 + x) * 24 + (r & 15)] = (_Float16)0.f;
    }
  }
  wait_async0();
  __syncthreads();

  // ---- S = Q K^T : wave owns 16 query rows x 384 keys (24 col-tiles) ----
  const int rowbase = wv * 16;
  v8f s[24];
#pragma unroll
  for (int jt = 0; jt < 24; ++jt) s[jt] = (v8f)0.f;
  v16h a0 = a_frag(Qs, QS_LD, rowbase + ml, 0, half);
  v16h a1 = a_frag(Qs, QS_LD, rowbase + ml, 32, half);
#pragma unroll
  for (int jc = 0; jc < 6; ++jc) {        // 4 col-tiles per chunk: load 8, wmma 8
    v16h bf[8];
#pragma unroll
    for (int u = 0; u < 4; ++u) {
      const int col = (jc * 4 + u) * 16 + ml;
      const _Float16* p0 = &Ksw[(0 + half) * KCH + col * 24];
      const _Float16* p1 = &Ksw[(2 + half) * KCH + col * 24];
      bf[u * 2]     = frag_cat(p0, p0 + 8);
      bf[u * 2 + 1] = frag_cat(p1, p1 + 8);
    }
#pragma unroll
    for (int u = 0; u < 4; ++u) {
      const int jt = jc * 4 + u;
      s[jt] = WMMA_F16(a0, bf[u * 2],     s[jt]);
      s[jt] = WMMA_F16(a1, bf[u * 2 + 1], s[jt]);
    }
  }

  // ---- exact-windowsize mask + f32 softmax ----
  float rmax[8], rsum[8];
#pragma unroll
  for (int r = 0; r < 8; ++r) rmax[r] = -3.0e38f;
#pragma unroll
  for (int jt = 0; jt < 24; ++jt) {
    int kpos = krow0 + jt * 16 + ml;
    bool kvalid = (unsigned)kpos < (unsigned)NSEQ;
#pragma unroll
    for (int r = 0; r < 8; ++r) {
      int qpos = qrow0 + rowbase + half * 8 + r;
      int diff = qpos - kpos;
      if (!kvalid || diff > WIN || diff < -WIN) s[jt][r] = -3.0e38f;
      rmax[r] = fmaxf(rmax[r], s[jt][r]);
    }
  }
#pragma unroll
  for (int off = 1; off < 16; off <<= 1)
#pragma unroll
    for (int r = 0; r < 8; ++r) rmax[r] = fmaxf(rmax[r], __shfl_xor(rmax[r], off, 32));
#pragma unroll
  for (int r = 0; r < 8; ++r) rsum[r] = 0.f;
#pragma unroll
  for (int jt = 0; jt < 24; ++jt)
#pragma unroll
    for (int r = 0; r < 8; ++r) {
      float p = __expf(s[jt][r] - rmax[r]);
      s[jt][r] = p;
      rsum[r] += p;
    }
#pragma unroll
  for (int off = 1; off < 16; off <<= 1)
#pragma unroll
    for (int r = 0; r < 8; ++r) rsum[r] += __shfl_xor(rsum[r], off, 32);

  // ---- P -> per-wave LDS (C-layout -> A-layout re-distribution) ----
  _Float16* Pw = Ps + wv * 16 * PS_LD;
#pragma unroll
  for (int jt = 0; jt < 24; ++jt)
#pragma unroll
    for (int r = 0; r < 8; ++r)
      Pw[(half * 8 + r) * PS_LD + jt * 16 + ml] = (_Float16)s[jt][r];
  __syncthreads();

  // ---- O = P V : 12 k-steps over 384 keys, 4 col-tiles of head dim ----
  v8f o[4];
#pragma unroll
  for (int ct = 0; ct < 4; ++ct) o[ct] = (v8f)0.f;
#pragma unroll
  for (int ks = 0; ks < 12; ++ks) {
    v16h a = a_frag(Pw, PS_LD, ml, ks * 32, half);
    v16h bf[4];
#pragma unroll
    for (int ct = 0; ct < 4; ++ct) {
      const _Float16* pv = &Vsw[(ks * 2 + half) * VCH + (ct * 16 + ml) * 24];
      bf[ct] = frag_cat(pv, pv + 8);
    }
#pragma unroll
    for (int ct = 0; ct < 4; ++ct) o[ct] = WMMA_F16(a, bf[ct], o[ct]);
  }
  float inv[8];
#pragma unroll
  for (int r = 0; r < 8; ++r) inv[r] = 1.0f / rsum[r];
#pragma unroll
  for (int ct = 0; ct < 4; ++ct)
#pragma unroll
    for (int r = 0; r < 8; ++r) {
      int row = qrow0 + rowbase + half * 8 + r;
      oh[(size_t)(b * NSEQ + row) * INNER + h * DHEAD + ct * 16 + ml] =
          (_Float16)(o[ct][r] * inv[r]);
    }
}

// ================= Kernel 3: output projection =================
__global__ __launch_bounds__(256)
void out_gemm(const _Float16* __restrict__ Ah, const float* __restrict__ Wout,
              float* __restrict__ out) {
  __shared__ _Float16 As[2][TM * A_LD];
  __shared__ _Float16 Bs[2][BSW_SZ];
  const int t = threadIdx.x;
  const int wv = t >> 5, lane = t & 31, half = lane >> 4, ml = lane & 15;
  const int m0 = blockIdx.y * TM;
  const int n0 = blockIdx.x * TN;
  const int NK = INNER / TK;

  v8f acc[8];
#pragma unroll
  for (int i = 0; i < 8; ++i) acc[i] = (v8f)0.f;

  stage_a_f16_async(As[0], Ah, INNER, m0, 0, t);
  stage_b_f32(Bs[0], Wout, DIM, 0, n0, t);
  wait_async0();
  __syncthreads();

  for (int kt = 0; kt < NK; ++kt) {
    const int cur = kt & 1;
    if (kt + 1 < NK) {
      stage_a_f16_async(As[cur ^ 1], Ah, INNER, m0, (kt + 1) * TK, t);
      stage_b_f32(Bs[cur ^ 1], Wout, DIM, (kt + 1) * TK, n0, t);
      if (kt + 2 < NK) {
        __builtin_prefetch(Ah + (size_t)(m0 + (t >> 1)) * INNER + (kt + 2) * TK, 0, 3);
        __builtin_prefetch(Wout + (size_t)((kt + 2) * TK + (t & 31)) * DIM + n0, 0, 3);
      }
    }
    v16h a = a_frag(As[cur], A_LD, wv * 16 + ml, 0, half);
    v16h bf[8];
#pragma unroll
    for (int nt = 0; nt < 8; ++nt) {
      const _Float16* bp = &Bs[cur][half * BSW_CH + (nt * 16 + ml) * BSW_NLD];
      bf[nt] = frag_cat(bp, bp + 8);
    }
#pragma unroll
    for (int nt = 0; nt < 8; ++nt) acc[nt] = WMMA_F16(a, bf[nt], acc[nt]);
    wait_async0();                 // async A-tile for next iter must land pre-barrier
    __syncthreads();
  }

#pragma unroll
  for (int nt = 0; nt < 8; ++nt) {
    int col = n0 + nt * 16 + ml;
#pragma unroll
    for (int r = 0; r < 8; ++r) {
      int grow = m0 + wv * 16 + r + half * 8;
      out[(size_t)grow * DIM + col] = acc[nt][r];
    }
  }
}

// ================= host launcher =================
extern "C" void kernel_launch(void* const* d_in, const int* in_sizes, int n_in,
                              void* d_out, int out_size, void* d_ws, size_t ws_size,
                              hipStream_t stream) {
  const float* x    = (const float*)d_in[0];   // (4, 8192, 512)
  const float* wqkv = (const float*)d_in[1];   // (512, 1536)
  const float* wout = (const float*)d_in[2];   // (512, 512)
  float* out = (float*)d_out;                  // (4, 8192, 512)

  const size_t per = (size_t)BATCH * HEADS * NSEQ * DHEAD;
  _Float16* qh = (_Float16*)d_ws;
  _Float16* kh = qh + per;
  _Float16* vh = kh + per;
  _Float16* oh = vh + per;                     // attn output, [32768][512] f16

  qkv_gemm<<<dim3((3 * INNER) / TN, MTOT / TM), 256, 0, stream>>>(x, wqkv, qh, kh, vh);

  const size_t smem = (size_t)SMEM_HALVES * sizeof(_Float16);  // ~260 KB (< 320 KB/WGP)
  local_attn<<<dim3(NW, HEADS, BATCH), 256, smem, stream>>>(qh, kh, vh, oh);

  out_gemm<<<dim3(DIM / TN, MTOT / TM), 256, 0, stream>>>(oh, wout, out);
}
